// SimLoss_12352325944210
// MI455X (gfx1250) — matrix operand order repaired and checked by reading, not compiled, not run on hardware
//
#include <hip/hip_runtime.h>

// ---------------------------------------------------------------------------
// SimLoss on MI455X (gfx1250, wave32, WMMA).
//  K1: per-batch 3x3 transform composition (+ zero accumulators)
//  K2: bilinear warp + L2-normalize + f32->f16 pack (one wave per pixel)
//  K3: streaming GEMM + logsumexp via V_WMMA_F32_16X16X32_F16,
//      register ping-pong B tiles (2 tiles/iter, no rotation movs),
//      diagonal recomputed in epilogue
//  K4: final scalar division
// ---------------------------------------------------------------------------

typedef _Float16 f16x8  __attribute__((ext_vector_type(8)));
typedef _Float16 f16x16 __attribute__((ext_vector_type(16)));
typedef float    f32x8  __attribute__((ext_vector_type(8)));

#define B_SZ 16
#define D_SZ 128
#define H_SZ 48
#define W_SZ 48
#define N_SZ (H_SZ * W_SZ)   // 2304
#define NT   (N_SZ / 16)     // 144 column/row tiles
#define TEMP_INV (1.0f / 0.07f)
#define LOG2E    1.4426950408889634f

// -------------------------------- K1 ---------------------------------------
__global__ void prep_kernel(const float* __restrict__ Hs_a,
                            const float* __restrict__ Hs_b,
                            const float* __restrict__ M_ab,
                            float* __restrict__ Tout,
                            float* __restrict__ accum) {
    int b = threadIdx.x;
    if (b == 0) { accum[0] = 0.0f; accum[1] = 0.0f; }
    if (b >= B_SZ) return;

    const float* A = Hs_a + b * 9;
    float a0=A[0],a1=A[1],a2=A[2],a3=A[3],a4=A[4],a5=A[5],a6=A[6],a7=A[7],a8=A[8];
    float det = a0*(a4*a8 - a5*a7) - a1*(a3*a8 - a5*a6) + a2*(a3*a7 - a4*a6);
    float id = 1.0f / det;
    float I[9];
    I[0]=(a4*a8-a5*a7)*id; I[1]=(a2*a7-a1*a8)*id; I[2]=(a1*a5-a2*a4)*id;
    I[3]=(a5*a6-a3*a8)*id; I[4]=(a0*a8-a2*a6)*id; I[5]=(a2*a3-a0*a5)*id;
    I[6]=(a3*a7-a4*a6)*id; I[7]=(a1*a6-a0*a7)*id; I[8]=(a0*a4-a1*a3)*id;

    float M3[9] = { M_ab[0], M_ab[1], M_ab[2],
                    M_ab[3], M_ab[4], M_ab[5],
                    0.0f,    0.0f,    1.0f };
    const float* Hb = Hs_b + b * 9;

    float P[9], Q[9];
    #pragma unroll
    for (int i = 0; i < 3; ++i)
        #pragma unroll
        for (int j = 0; j < 3; ++j) {
            float s = 0.0f;
            #pragma unroll
            for (int k = 0; k < 3; ++k) s += Hb[i*3+k] * M3[k*3+j];
            P[i*3+j] = s;
        }
    #pragma unroll
    for (int i = 0; i < 3; ++i)
        #pragma unroll
        for (int j = 0; j < 3; ++j) {
            float s = 0.0f;
            #pragma unroll
            for (int k = 0; k < 3; ++k) s += P[i*3+k] * I[k*3+j];
            Q[i*3+j] = s;
        }
    // T = S_inv @ Q @ S : rows 0,1 scaled by 1/16, cols 0,1 scaled by 16
    #pragma unroll
    for (int i = 0; i < 3; ++i)
        #pragma unroll
        for (int j = 0; j < 3; ++j) {
            float v = Q[i*3+j];
            if (i < 2) v *= (1.0f / 16.0f);
            if (j < 2) v *= 16.0f;
            Tout[b*9 + i*3 + j] = v;
        }
}

// -------------------------------- K2 ---------------------------------------
// One wave per pixel n of batch b; lanes split the 128 channels (4 per lane).
__global__ __launch_bounds__(256) void warp_norm_kernel(
    const float* __restrict__ feats_a, const float* __restrict__ feats_b,
    const float* __restrict__ T, _Float16* __restrict__ fa16,
    _Float16* __restrict__ wb16, float* __restrict__ maskOut) {

    int wave = threadIdx.x >> 5;
    int lane = threadIdx.x & 31;
    int p = blockIdx.x * 8 + wave;        // global pixel id in [0, B*N)
    int b = p / N_SZ;
    int n = p - b * N_SZ;
    float fx = (float)(n % W_SZ);
    float fy = (float)(n / W_SZ);

    const float* Tb = T + b * 9;
    float X = Tb[0]*fx + Tb[1]*fy + Tb[2];
    float Y = Tb[3]*fx + Tb[4]*fy + Tb[5];
    float Z = Tb[6]*fx + Tb[7]*fy + Tb[8] + 1e-8f;
    float xn = 2.0f * (X / Z) / (float)(W_SZ - 1) - 1.0f;
    float yn = 2.0f * (Y / Z) / (float)(H_SZ - 1) - 1.0f;
    float valid = (xn >= -1.0f && xn <= 1.0f && yn >= -1.0f && yn <= 1.0f) ? 1.0f : 0.0f;

    float ix = (xn + 1.0f) * 0.5f * (float)(W_SZ - 1);
    float iy = (yn + 1.0f) * 0.5f * (float)(H_SZ - 1);
    float x0f = floorf(ix), y0f = floorf(iy);
    float wx1 = ix - x0f, wx0 = 1.0f - wx1;
    float wy1 = iy - y0f, wy0 = 1.0f - wy1;
    int x0 = (int)x0f, y0 = (int)y0f;

    int   ti[4];
    float tw[4];
    int   xs[2]  = { x0, x0 + 1 };
    int   ys[2]  = { y0, y0 + 1 };
    float wxs[2] = { wx0, wx1 };
    float wys[2] = { wy0, wy1 };
    #pragma unroll
    for (int ty = 0; ty < 2; ++ty)
        #pragma unroll
        for (int tx = 0; tx < 2; ++tx) {
            int xi = xs[tx], yi = ys[ty];
            float tv = (xi >= 0 && xi <= W_SZ-1 && yi >= 0 && yi <= H_SZ-1) ? 1.0f : 0.0f;
            int xc = min(max(xi, 0), W_SZ - 1);
            int yc = min(max(yi, 0), H_SZ - 1);
            ti[ty*2+tx] = yc * W_SZ + xc;
            tw[ty*2+tx] = wxs[tx] * wys[ty] * tv;   // zero-padded taps
        }

    float va[4], vb[4];
    float ssa = 0.0f, ssb = 0.0f;
    #pragma unroll
    for (int c = 0; c < 4; ++c) {
        int d = lane + 32 * c;
        const float* fb = feats_b + (size_t)(b * D_SZ + d) * N_SZ;
        float acc = tw[0]*fb[ti[0]] + tw[1]*fb[ti[1]]
                  + tw[2]*fb[ti[2]] + tw[3]*fb[ti[3]];
        vb[c] = acc; ssb += acc * acc;
        float av = feats_a[(size_t)(b * D_SZ + d) * N_SZ + n];
        va[c] = av; ssa += av * av;
    }
    #pragma unroll
    for (int m = 16; m >= 1; m >>= 1) {
        ssa += __shfl_xor(ssa, m, 32);
        ssb += __shfl_xor(ssb, m, 32);
    }
    float inva = 1.0f / fmaxf(sqrtf(ssa), 1e-12f);
    float invb = 1.0f / fmaxf(sqrtf(ssb), 1e-12f);

    _Float16* far = fa16 + (size_t)p * D_SZ;
    _Float16* wbr = wb16 + (size_t)p * D_SZ;
    #pragma unroll
    for (int c = 0; c < 4; ++c) {
        int d = lane + 32 * c;
        far[d] = (_Float16)(va[c] * inva);
        wbr[d] = (_Float16)(vb[c] * invb);
    }
    if (lane == 0) maskOut[p] = valid;
}

// -------------------------------- K3 ---------------------------------------
// One block per (batch, 16-row strip). 8 waves stride the 144 column tiles
// (exactly 18 each -> uniform trip count). Two register buffers ping-pong:
// each outer iteration processes 2 tiles and prefetches 2 tiles ahead, so
// no buffer-rotation moves are needed and loads hide behind WMMA+exp work.
// |logit| <= 1/TEMP ~= 14.3  =>  exp never overflows f32; no max tracking.
// Diagonal logit is recomputed in the epilogue (dot over the same f16 data).
__global__ __launch_bounds__(256) void gemm_lse_kernel(
    const _Float16* __restrict__ fa16, const _Float16* __restrict__ wb16,
    const float* __restrict__ maskIn, float* __restrict__ accum) {

    int blk = blockIdx.x;
    int b  = blk / NT;
    int rt = blk - b * NT;
    int tid = threadIdx.x;
    int wave = tid >> 5;
    int lane = tid & 31;
    int lane15 = lane & 15;
    int hi = lane >> 4;

    __shared__ float s_sums[8][16];

    // A fragments (16x32 f16, lane=M; low lanes K {0..15},
    // high lanes K {16..31}) -- loaded once, reused for 18 tiles.
    const _Float16* fa_row = fa16 + (size_t)(b * N_SZ + rt * 16 + lane15) * D_SZ;
    f16x16 afrag[4];
    #pragma unroll
    for (int kb = 0; kb < 4; ++kb) {
        int d0 = kb * 32 + hi * 16;
        f16x8 lo = *(const f16x8*)(fa_row + d0);
        f16x8 hh = *(const f16x8*)(fa_row + d0 + 8);
        afrag[kb] = __builtin_shufflevector(lo, hh,
                        0,1,2,3,4,5,6,7,8,9,10,11,12,13,14,15);
    }

    // B fragment (32x16 f16 K-major: lane=N, K contiguous per half-wave)
    // == one 32B chunk of row ct*16+lane15 of row-major wb16, per lane.
    const _Float16* wb_lane =
        wb16 + ((size_t)b * N_SZ + lane15) * D_SZ + hi * 16;
    const size_t tileStride = (size_t)16 * D_SZ;   // halves per column tile

#define BFRAG(t, kb) (*(const f16x16*)(wb_lane + (size_t)(t) * tileStride + (kb) * 32))

    float s[8];
    #pragma unroll
    for (int r = 0; r < 8; ++r) s[r] = 0.0f;

    const float expScale = TEMP_INV * LOG2E;       // fold 1/TEMP into exp2
    const int ITER2 = NT / 16;                     // 9 outer iters x 2 tiles

    int ct = wave;                                 // this wave's tiles: wave+8k
    f16x16 b0[4], b1[4];
    #pragma unroll
    for (int kb = 0; kb < 4; ++kb) b0[kb] = BFRAG(ct, kb);
    #pragma unroll
    for (int kb = 0; kb < 4; ++kb) b1[kb] = BFRAG(ct + 8, kb);

    for (int it = 0; it < ITER2; ++it) {
        // prefetch targets 2 tiles ahead (clamped in-bounds on last iter)
        int t0n = (it + 1 < ITER2) ? (ct + 16) : wave;
        int t1n = (it + 1 < ITER2) ? (ct + 24) : wave;

        // ---- tile ct (buffer 0) ----
        f32x8 c0 = {};
        #pragma unroll
        for (int kb = 0; kb < 4; ++kb)
            c0 = __builtin_amdgcn_wmma_f32_16x16x32_f16(
                    false, afrag[kb], false, b0[kb], (short)0, c0, false, false);
        #pragma unroll
        for (int kb = 0; kb < 4; ++kb) b0[kb] = BFRAG(t0n, kb);   // refill
        #pragma unroll
        for (int r = 0; r < 8; ++r)
            s[r] += __builtin_amdgcn_exp2f(c0[r] * expScale);     // = exp(c/TEMP)

        // ---- tile ct+8 (buffer 1) ----
        f32x8 c1 = {};
        #pragma unroll
        for (int kb = 0; kb < 4; ++kb)
            c1 = __builtin_amdgcn_wmma_f32_16x16x32_f16(
                    false, afrag[kb], false, b1[kb], (short)0, c1, false, false);
        #pragma unroll
        for (int kb = 0; kb < 4; ++kb) b1[kb] = BFRAG(t1n, kb);   // refill
        #pragma unroll
        for (int r = 0; r < 8; ++r)
            s[r] += __builtin_amdgcn_exp2f(c1[r] * expScale);

        ct += 16;
    }
#undef BFRAG

    // reduce the 16 column-lanes of each half-wave (rows 0..7 / 8..15)
    #pragma unroll
    for (int r = 0; r < 8; ++r) {
        float v = s[r];
        v += __shfl_xor(v, 1, 32);
        v += __shfl_xor(v, 2, 32);
        v += __shfl_xor(v, 4, 32);
        v += __shfl_xor(v, 8, 32);
        if (lane15 == 0) s_sums[wave][r + hi * 8] = v;
    }
    __syncthreads();

    if (tid < 16) {
        float tot = 0.0f;
        #pragma unroll
        for (int w = 0; w < 8; ++w) tot += s_sums[w][tid];

        // diagonal logit: dot over the identical f16 operands (bit-consistent
        // with the WMMA inputs), 128 FMAs per epilogue thread.
        int row = b * N_SZ + rt * 16 + tid;
        const _Float16* ar = fa16 + (size_t)row * D_SZ;
        const _Float16* br = wb16 + (size_t)row * D_SZ;
        float dacc = 0.0f;
        #pragma unroll
        for (int k = 0; k < D_SZ / 8; ++k) {
            f16x8 av = *(const f16x8*)(ar + k * 8);
            f16x8 bv = *(const f16x8*)(br + k * 8);
            #pragma unroll
            for (int e = 0; e < 8; ++e)
                dacc += (float)av[e] * (float)bv[e];
        }

        float m = maskIn[(size_t)b * N_SZ + rt * 16 + tid];
        float loss = (__logf(tot) - dacc * TEMP_INV) * m;
        #pragma unroll
        for (int mm = 8; mm >= 1; mm >>= 1) {
            loss += __shfl_xor(loss, mm, 32);
            m    += __shfl_xor(m,    mm, 32);
        }
        if (tid == 0) {
            atomicAdd(&accum[0], loss);
            atomicAdd(&accum[1], m);
        }
    }
}

// -------------------------------- K4 ---------------------------------------
__global__ void finalize_kernel(const float* __restrict__ accum,
                                float* __restrict__ out) {
    float total = accum[0], nvalid = accum[1];
    out[0] = (nvalid > 0.0f) ? (total / fmaxf(nvalid, 1.0f)) : 0.0f;
}

// ---------------------------------------------------------------------------
extern "C" void kernel_launch(void* const* d_in, const int* in_sizes, int n_in,
                              void* d_out, int out_size, void* d_ws, size_t ws_size,
                              hipStream_t stream) {
    (void)in_sizes; (void)n_in; (void)out_size; (void)ws_size;
    const float* feats_a = (const float*)d_in[0];
    const float* feats_b = (const float*)d_in[1];
    const float* Hs_a    = (const float*)d_in[2];
    const float* Hs_b    = (const float*)d_in[3];
    const float* M_ab    = (const float*)d_in[4];

    char* ws = (char*)d_ws;
    float*    T     = (float*)ws;                         //   576 B
    float*    accum = (float*)(ws + 640);                 //     8 B
    float*    maskb = (float*)(ws + 1024);                // 147456 B
    _Float16* fa16  = (_Float16*)(ws + 1024 + 147456);    // 9437184 B (32B aligned)
    _Float16* wb16  = (_Float16*)(ws + 1024 + 147456 + 9437184);

    prep_kernel<<<1, 32, 0, stream>>>(Hs_a, Hs_b, M_ab, T, accum);
    warp_norm_kernel<<<(B_SZ * N_SZ) / 8, 256, 0, stream>>>(
        feats_a, feats_b, T, fa16, wb16, maskb);
    gemm_lse_kernel<<<B_SZ * NT, 256, 0, stream>>>(fa16, wb16, maskb, accum);
    finalize_kernel<<<1, 1, 0, stream>>>(accum, (float*)d_out);
}
